// CausalChunkedPrefill_64888365908146
// MI455X (gfx1250) — matrix operand order repaired and checked by reading, not compiled, not run on hardware
//
#include <hip/hip_runtime.h>
#include <stdint.h>
#include <stddef.h>

// Problem constants (from reference): B=2, S=2048, D=1024, H=16, hd=64
#define B_DIM 2
#define S_DIM 2048
#define D_DIM 1024
#define H_DIM 16
#define HD_DIM 64
#define M_ROWS (B_DIM * S_DIM) // 4096

typedef __attribute__((ext_vector_type(16))) __bf16 v16bf;
typedef __attribute__((ext_vector_type(8)))  __bf16 v8bf;
typedef __attribute__((ext_vector_type(8)))  float  v8f;

// ---------- helpers ----------

__device__ inline v8f vzero() {
  v8f v;
#pragma unroll
  for (int i = 0; i < 8; ++i) v[i] = 0.0f;
  return v;
}

// fp32 -> bf16 round-to-nearest-even (bit manipulation; no bf16 VALU needed)
__device__ inline __bf16 f2bf(float f) {
  uint32_t u = __builtin_bit_cast(uint32_t, f);
  uint32_t r = u + 0x7FFFu + ((u >> 16) & 1u);
  uint16_t h = (uint16_t)(r >> 16);
  return __builtin_bit_cast(__bf16, h);
}

// A-matrix 16x32 bf16 fragment (ISA 7.12.2):
// lanes 0-15: M=lane, K=0..7 (v0-3) and K=16..23 (v4-7)
// lanes 16-31: M=lane-16, K=8..15 and K=24..31
__device__ inline v16bf ld_afrag(const __bf16* p, int ld, int lane) {
  const int m  = lane & 15;
  const int kh = (lane >> 4) << 3; // 0 or 8
  const __bf16* r = p + m * ld + kh;
  v8bf lo = *(const v8bf*)(r);        // K = kh .. kh+7
  v8bf hi = *(const v8bf*)(r + 16);   // K = kh+16 .. kh+23
  return __builtin_shufflevector(lo, hi, 0,1,2,3,4,5,6,7,8,9,10,11,12,13,14,15);
}

// B-matrix 32x16 bf16 fragment: lanes 0-15: N=lane, K=0..7 and K=16..23; lanes 16-31: N=lane-16, K=8..15 and K=24..31
// Source rows are laid out [N][K] (i.e. W row-major, or K/V rows), contiguous in K.
__device__ inline v16bf ld_bfrag(const __bf16* p, int ld, int lane) {
  const int n  = lane & 15;
  const int k0 = (lane >> 4) << 3; // 0 or 8
  const __bf16* r = p + n * ld + k0;
  v8bf lo = *(const v8bf*)(r);
  v8bf hi = *(const v8bf*)(r + 16);
  return __builtin_shufflevector(lo, hi, 0,1,2,3,4,5,6,7,8,9,10,11,12,13,14,15);
}

__device__ inline v8f wmma_bf16(v16bf a, v16bf b, v8f c) {
  return __builtin_amdgcn_wmma_f32_16x16x32_bf16(false, a, false, b, (short)0, c,
                                                 false, false);
}

// ---------- fp32 -> bf16 conversion ----------

__global__ void cvt_bf16_kernel(const float* __restrict__ src,
                                __bf16* __restrict__ dst, int n) {
  int i = blockIdx.x * blockDim.x + threadIdx.x;
  const int stride = gridDim.x * blockDim.x;
  for (; i < n; i += stride) dst[i] = f2bf(src[i]);
}

// ---------- shared GEMM core: C[64x64] += A[64xK] * W[64xK]^T ----------
// 128 threads = 4 waves (2x2), each wave owns a 32x32 tile = 2x2 WMMA frags.

#define GM 64
#define GN 64
#define GK 32
#define GLD 40 // LDS row stride in elements (32 + 8 pad, keeps 16B alignment)

__device__ inline void gemm_core(const __bf16* __restrict__ A,
                                 const __bf16* __restrict__ W,
                                 __bf16* lds, int M0, int N0, int K,
                                 v8f c[2][2], int tid) {
  __bf16* As = lds;
  __bf16* Bs = lds + GM * GLD;
  const int lane = tid & 31;
  const int wave = tid >> 5;
  const int wm = (wave >> 1) * 32;
  const int wn = (wave & 1) * 32;

  for (int kb = 0; kb < K; kb += GK) {
    __syncthreads();
    {
      const int row = tid >> 1;          // 0..63
      const int seg = (tid & 1) * 16;    // 0 or 16 elements
      const __bf16* ag = A + (size_t)(M0 + row) * K + kb + seg;
      int4 a0 = *(const int4*)(ag);
      int4 a1 = *(const int4*)(ag + 8);
      const __bf16* bg = W + (size_t)(N0 + row) * K + kb + seg;
      int4 b0 = *(const int4*)(bg);
      int4 b1 = *(const int4*)(bg + 8);
      __bf16* ad = As + row * GLD + seg;
      *(int4*)(ad) = a0; *(int4*)(ad + 8) = a1;
      __bf16* bd = Bs + row * GLD + seg;
      *(int4*)(bd) = b0; *(int4*)(bd + 8) = b1;
    }
    __syncthreads();

    v16bf af[2], bfv[2];
#pragma unroll
    for (int i = 0; i < 2; ++i) af[i]  = ld_afrag(As + (wm + i * 16) * GLD, GLD, lane);
#pragma unroll
    for (int j = 0; j < 2; ++j) bfv[j] = ld_bfrag(Bs + (wn + j * 16) * GLD, GLD, lane);
#pragma unroll
    for (int i = 0; i < 2; ++i)
#pragma unroll
      for (int j = 0; j < 2; ++j)
        c[i][j] = wmma_bf16(af[i], bfv[j], c[i][j]);
  }
}

// ---------- QKV projection: y = x @ W^T + b, stored in attention layouts ----------
// z=0 -> Q[b,h,s,64], z=1 -> K[b,h,s,64], z=2 -> V^T [b,h,64,s]

__global__ __launch_bounds__(128) void qkv_gemm_kernel(
    const __bf16* __restrict__ xh,
    const __bf16* __restrict__ Wq, const __bf16* __restrict__ Wk,
    const __bf16* __restrict__ Wv,
    const float* __restrict__ bq, const float* __restrict__ bk,
    const float* __restrict__ bv,
    __bf16* __restrict__ Qb, __bf16* __restrict__ Kb, __bf16* __restrict__ Vtb) {
  __shared__ __bf16 lds[2 * GM * GLD];
  const int z = blockIdx.z;
  const __bf16* W    = (z == 0) ? Wq : (z == 1) ? Wk : Wv;
  const float*  bias = (z == 0) ? bq : (z == 1) ? bk : bv;
  const int M0 = blockIdx.y * GM;
  const int N0 = blockIdx.x * GN;

  v8f c[2][2];
#pragma unroll
  for (int i = 0; i < 2; ++i)
#pragma unroll
    for (int j = 0; j < 2; ++j) c[i][j] = vzero();

  gemm_core(xh, W, lds, M0, N0, D_DIM, c, threadIdx.x);

  const int lane = threadIdx.x & 31;
  const int wave = threadIdx.x >> 5;
  const int wm = (wave >> 1) * 32, wn = (wave & 1) * 32;
  const int rowoff = (lane >> 4) << 3;
  const int nl = lane & 15;

#pragma unroll
  for (int j = 0; j < 2; ++j) {
    const int n = N0 + wn + j * 16 + nl;
    const float bn = bias[n];
    const int h = n >> 6, d = n & 63;
#pragma unroll
    for (int i = 0; i < 2; ++i) {
#pragma unroll
      for (int r = 0; r < 8; ++r) {
        const int m = M0 + wm + i * 16 + rowoff + r;
        const int b = m >> 11;            // m / S_DIM
        const int s = m & (S_DIM - 1);
        const __bf16 hv = f2bf(c[i][j][r] + bn);
        if (z < 2) {
          __bf16* dst = (z == 0) ? Qb : Kb;
          dst[(((size_t)(b * H_DIM + h)) * S_DIM + s) * HD_DIM + d] = hv;
        } else {
          Vtb[(((size_t)(b * H_DIM + h)) * HD_DIM + d) * S_DIM + s] = hv;
        }
      }
    }
  }
}

// ---------- Output projection: out = ctx @ Wo^T + bo (fp32 out) ----------

__global__ __launch_bounds__(128) void out_gemm_kernel(
    const __bf16* __restrict__ ctxh, const __bf16* __restrict__ Wo,
    const float* __restrict__ bo, float* __restrict__ out) {
  __shared__ __bf16 lds[2 * GM * GLD];
  const int M0 = blockIdx.y * GM;
  const int N0 = blockIdx.x * GN;

  v8f c[2][2];
#pragma unroll
  for (int i = 0; i < 2; ++i)
#pragma unroll
    for (int j = 0; j < 2; ++j) c[i][j] = vzero();

  gemm_core(ctxh, Wo, lds, M0, N0, D_DIM, c, threadIdx.x);

  const int lane = threadIdx.x & 31;
  const int wave = threadIdx.x >> 5;
  const int wm = (wave >> 1) * 32, wn = (wave & 1) * 32;
  const int rowoff = (lane >> 4) << 3;
  const int nl = lane & 15;

#pragma unroll
  for (int j = 0; j < 2; ++j) {
    const int n = N0 + wn + j * 16 + nl;
    const float bn = bo[n];
#pragma unroll
    for (int i = 0; i < 2; ++i) {
#pragma unroll
      for (int r = 0; r < 8; ++r) {
        const int m = M0 + wm + i * 16 + rowoff + r;
        out[(size_t)m * D_DIM + n] = c[i][j][r] + bn;
      }
    }
  }
}

// ---------- Flash-style causal attention ----------
// Block = 4 waves, 64 query rows per block (16 per wave); key blocks of 64.
// LDS: K tile [64 keys][64 d], V^T tile [64 d][64 keys], per-wave P staging [16][64].

#define ALD 72 // padded LDS stride (144B rows, 16B-aligned offsets)

__global__ __launch_bounds__(128) void attn_kernel(
    const __bf16* __restrict__ Qb, const __bf16* __restrict__ Kb,
    const __bf16* __restrict__ Vtb, __bf16* __restrict__ ctxh) {
  __shared__ __bf16 lds[64 * ALD * 2 + 4 * 16 * ALD];
  __bf16* Ks = lds;
  __bf16* Vs = lds + 64 * ALD;

  const int tid = threadIdx.x, lane = tid & 31, wave = tid >> 5;
  __bf16* Ps = lds + 2 * 64 * ALD + wave * (16 * ALD);

  const int bh = blockIdx.y; // b*H + h
  const int qb = blockIdx.x; // query block of 64 rows
  const __bf16* Qp = Qb  + (size_t)bh * S_DIM * HD_DIM;
  const __bf16* Kp = Kb  + (size_t)bh * S_DIM * HD_DIM;
  const __bf16* Vp = Vtb + (size_t)bh * HD_DIM * S_DIM;

  const int qr0 = qb * 64 + wave * 16;
  const int rowoff = (lane >> 4) << 3;
  const int nl = lane & 15;

  // Q fragments stay resident (16 rows x 64 d = 2 A-frags)
  v16bf qf[2];
#pragma unroll
  for (int kk = 0; kk < 2; ++kk)
    qf[kk] = ld_afrag(Qp + (size_t)qr0 * HD_DIM + kk * 32, HD_DIM, lane);

  v8f o[4];
#pragma unroll
  for (int i = 0; i < 4; ++i) o[i] = vzero();
  float mrow[8], lrow[8];
#pragma unroll
  for (int r = 0; r < 8; ++r) { mrow[r] = -1e30f; lrow[r] = 0.0f; }
  const float sc = 0.18033688f; // (1/sqrt(64)) * log2(e); exp2-domain softmax

  for (int jb = 0; jb <= qb; ++jb) {
    const int kb0 = jb * 64;
    __syncthreads();
    {
      const int row = tid >> 1;         // 0..63
      const int seg = (tid & 1) * 32;   // 0 or 32 elements
      const __bf16* kg = Kp + (size_t)(kb0 + row) * HD_DIM + seg;
      int4 k0 = *(const int4*)(kg);
      int4 k1 = *(const int4*)(kg + 8);
      int4 k2 = *(const int4*)(kg + 16);
      int4 k3 = *(const int4*)(kg + 24);
      const __bf16* vg = Vp + (size_t)row * S_DIM + kb0 + seg;
      int4 v0 = *(const int4*)(vg);
      int4 v1 = *(const int4*)(vg + 8);
      int4 v2 = *(const int4*)(vg + 16);
      int4 v3 = *(const int4*)(vg + 24);
      __bf16* kd = Ks + row * ALD + seg;
      *(int4*)(kd) = k0; *(int4*)(kd + 8) = k1;
      *(int4*)(kd + 16) = k2; *(int4*)(kd + 24) = k3;
      __bf16* vd = Vs + row * ALD + seg;
      *(int4*)(vd) = v0; *(int4*)(vd + 8) = v1;
      *(int4*)(vd + 16) = v2; *(int4*)(vd + 24) = v3;
    }
    __syncthreads();

    // scores: 16 q-rows x 64 keys = 4 C-frags, each 2 WMMA k-steps over d
    float sv[4][8];
    const bool diag = (jb == qb);
#pragma unroll
    for (int j2 = 0; j2 < 4; ++j2) {
      v8f s = vzero();
#pragma unroll
      for (int kk = 0; kk < 2; ++kk) {
        v16bf kf = ld_bfrag(Ks + (j2 * 16) * ALD + kk * 32, ALD, lane);
        s = wmma_bf16(qf[kk], kf, s);
      }
      const int key = kb0 + j2 * 16 + nl;
#pragma unroll
      for (int r = 0; r < 8; ++r) {
        float v = s[r] * sc;
        if (diag && (key > qr0 + rowoff + r)) v = -1e30f;
        sv[j2][r] = v;
      }
    }

    // online softmax (half-wave reductions: xor masks 1..8 stay within C-row groups)
#pragma unroll
    for (int r = 0; r < 8; ++r) {
      float vmax = fmaxf(fmaxf(sv[0][r], sv[1][r]), fmaxf(sv[2][r], sv[3][r]));
#pragma unroll
      for (int d = 8; d >= 1; d >>= 1) vmax = fmaxf(vmax, __shfl_xor(vmax, d, 32));
      const float mnew = fmaxf(mrow[r], vmax);
      const float alpha = exp2f(mrow[r] - mnew);
      mrow[r] = mnew;
      lrow[r] *= alpha;
      o[0][r] *= alpha; o[1][r] *= alpha; o[2][r] *= alpha; o[3][r] *= alpha;
      float rs = 0.0f;
#pragma unroll
      for (int j2 = 0; j2 < 4; ++j2) {
        const float p = exp2f(sv[j2][r] - mnew);
        sv[j2][r] = p;
        rs += p;
      }
#pragma unroll
      for (int d = 8; d >= 1; d >>= 1) rs += __shfl_xor(rs, d, 32);
      lrow[r] += rs;
    }

    // transpose P (C-layout f32) -> A-layout bf16 via wave-private LDS
#pragma unroll
    for (int j2 = 0; j2 < 4; ++j2)
#pragma unroll
      for (int r = 0; r < 8; ++r)
        Ps[(rowoff + r) * ALD + j2 * 16 + nl] = f2bf(sv[j2][r]);
    __syncthreads();

    // O += P @ V : 4 d-frags x 2 WMMA k-steps over keys
#pragma unroll
    for (int c4 = 0; c4 < 4; ++c4) {
#pragma unroll
      for (int kk = 0; kk < 2; ++kk) {
        v16bf pa = ld_afrag(Ps + kk * 32, ALD, lane);
        v16bf vb = ld_bfrag(Vs + (c4 * 16) * ALD + kk * 32, ALD, lane);
        o[c4] = wmma_bf16(pa, vb, o[c4]);
      }
    }
  }

  // finalize: divide by l, store ctx as [B,S,D] bf16 for the output projection
  const int b = bh >> 4, h = bh & 15;
#pragma unroll
  for (int c4 = 0; c4 < 4; ++c4) {
#pragma unroll
    for (int r = 0; r < 8; ++r) {
      const float val = o[c4][r] / lrow[r];
      const int srow = qr0 + rowoff + r;
      const int d = c4 * 16 + nl;
      ctxh[((size_t)b * S_DIM + srow) * D_DIM + h * HD_DIM + d] = f2bf(val);
    }
  }
}

// ---------- host ----------

extern "C" void kernel_launch(void* const* d_in, const int* in_sizes, int n_in,
                              void* d_out, int out_size, void* d_ws, size_t ws_size,
                              hipStream_t stream) {
  (void)in_sizes; (void)n_in; (void)out_size; (void)ws_size;
  const float* x  = (const float*)d_in[0];
  const float* Wq = (const float*)d_in[1];
  const float* bq = (const float*)d_in[2];
  const float* Wk = (const float*)d_in[3];
  const float* bk = (const float*)d_in[4];
  const float* Wv = (const float*)d_in[5];
  const float* bv = (const float*)d_in[6];
  const float* Wo = (const float*)d_in[7];
  const float* bo = (const float*)d_in[8];
  float* out = (float*)d_out;

  __bf16* ws = (__bf16*)d_ws;
  size_t off = 0;
  __bf16* xh   = ws + off; off += (size_t)M_ROWS * D_DIM;                 // 4096x1024
  __bf16* wqh  = ws + off; off += (size_t)D_DIM * D_DIM;
  __bf16* wkh  = ws + off; off += (size_t)D_DIM * D_DIM;
  __bf16* wvh  = ws + off; off += (size_t)D_DIM * D_DIM;
  __bf16* woh  = ws + off; off += (size_t)D_DIM * D_DIM;
  __bf16* Qbuf = ws + off; off += (size_t)B_DIM * H_DIM * S_DIM * HD_DIM;
  __bf16* Kbuf = ws + off; off += (size_t)B_DIM * H_DIM * S_DIM * HD_DIM;
  __bf16* Vtbf = ws + off; off += (size_t)B_DIM * H_DIM * HD_DIM * S_DIM;
  __bf16* ctxh = ws + off; off += (size_t)M_ROWS * D_DIM;

  // fp32 -> bf16
  cvt_bf16_kernel<<<2048, 256, 0, stream>>>(x,  xh,  M_ROWS * D_DIM);
  cvt_bf16_kernel<<<1024, 256, 0, stream>>>(Wq, wqh, D_DIM * D_DIM);
  cvt_bf16_kernel<<<1024, 256, 0, stream>>>(Wk, wkh, D_DIM * D_DIM);
  cvt_bf16_kernel<<<1024, 256, 0, stream>>>(Wv, wvh, D_DIM * D_DIM);
  cvt_bf16_kernel<<<1024, 256, 0, stream>>>(Wo, woh, D_DIM * D_DIM);

  // fused QKV projection (z selects W); writes Q,K row-major and V transposed
  qkv_gemm_kernel<<<dim3(D_DIM / GN, M_ROWS / GM, 3), 128, 0, stream>>>(
      xh, wqh, wkh, wvh, bq, bk, bv, Qbuf, Kbuf, Vtbf);

  // flash-style causal attention
  attn_kernel<<<dim3(S_DIM / 64, B_DIM * H_DIM), 128, 0, stream>>>(
      Qbuf, Kbuf, Vtbf, ctxh);

  // output projection -> fp32
  out_gemm_kernel<<<dim3(D_DIM / GN, M_ROWS / GM), 128, 0, stream>>>(
      ctxh, woh, bo, out);
}
